// Seq2SeqDeltaModel_89893665505363
// MI455X (gfx1250) — matrix operand order repaired and checked by reading, not compiled, not run on hardware
//
#include <hip/hip_runtime.h>
#include <hip/hip_bf16.h>
#include <stdint.h>
#include <stddef.h>

typedef unsigned short ushort_t;
typedef __attribute__((ext_vector_type(16))) __bf16 v16bf;
typedef __attribute__((ext_vector_type(8)))  float  v8f;
typedef __attribute__((ext_vector_type(4)))  int    v4i;

#define B_SZ   256
#define T_SZ   512
#define H_SZ   128
#define G4H    512   // 4*H
#define NDYN_K 10
#define FSTEP  50
#define OUT_K  2
#define RPW    16    // batch rows per workgroup
#define NTHR   256   // 8 waves

// ---- CDNA5 async global->LDS (guarded; falls back to plain copy) ----
#if defined(__has_builtin)
#if __has_builtin(__builtin_amdgcn_global_load_async_to_lds_b128) && \
    __has_builtin(__builtin_amdgcn_s_wait_asynccnt)
#define USE_ASYNC_LDS 1
#endif
#endif
#ifndef USE_ASYNC_LDS
#define USE_ASYNC_LDS 0
#endif

__device__ __forceinline__ void async_cp16(const void* g, void* l) {
#if USE_ASYNC_LDS
  __builtin_amdgcn_global_load_async_to_lds_b128((v4i*)g, (v4i*)l, 0, 0);
#else
  *(int4*)l = *(const int4*)g;
#endif
}
__device__ __forceinline__ void wait_async0() {
#if USE_ASYNC_LDS
  __builtin_amdgcn_s_wait_asynccnt(0);
#endif
}
__device__ __forceinline__ void wait_async1() {
#if USE_ASYNC_LDS
  __builtin_amdgcn_s_wait_asynccnt(1);
#endif
}

// ---------------- helpers ----------------
__device__ __forceinline__ ushort_t f2bf(float f) {
  union { float f; unsigned u; } cv; cv.f = f;
  unsigned u = cv.u;
  unsigned r = (u + 0x7fffu + ((u >> 16) & 1u)) >> 16;
  return (ushort_t)r;
}
__device__ __forceinline__ float sigf(float x) { return 1.0f / (1.0f + expf(-x)); }

// A-fragment (16 x K, bf16, wave32 WMMA 16x16x32):
// within a 32-wide K chunk, element (row m, kk):
//   lane = m + 16*((kk>>4)&1),  e = kk&15
__device__ __forceinline__ void a_store(ushort_t* Afrag, int row, int k, ushort_t v) {
  int ck = k >> 5, kk = k & 31;
  int lane = row + 16 * ((kk >> 4) & 1);
  int e = kk & 15;
  Afrag[(ck * 32 + lane) * 16 + e] = v;
}

// B-fragment pack: element (kk, n) within chunk: lane = (n&15) + 16*((kk>>4)&1), e = kk&15
template<int NCHUNK>
__device__ __forceinline__ v8f wave_tile_gemm(const ushort_t* Afrag, const ushort_t* Wfrag,
                                              int tile, int lane, v8f acc) {
#pragma unroll
  for (int ck = 0; ck < NCHUNK; ++ck) {
    v16bf a = *(const v16bf*)(Afrag + (ck * 32 + lane) * 16);
    v16bf b = *(const v16bf*)(Wfrag + ((size_t)(tile * NCHUNK + ck) * 32 + lane) * 16);
    acc = __builtin_amdgcn_wmma_f32_16x16x32_bf16(false, a, false, b, (short)0, acc,
                                                  false, false);
  }
  return acc;
}

// load 4 gate octets for (row, jb) as float4 pairs
__device__ __forceinline__ void load_gates(const float* g_buf, int row, int jb,
                                           float* gi, float* gf, float* gg, float* go) {
  const float* gp = g_buf + row * G4H + jb;
  *(float4*)&gi[0] = *(const float4*)(gp);
  *(float4*)&gi[4] = *(const float4*)(gp + 4);
  *(float4*)&gf[0] = *(const float4*)(gp + H_SZ);
  *(float4*)&gf[4] = *(const float4*)(gp + H_SZ + 4);
  *(float4*)&gg[0] = *(const float4*)(gp + 2 * H_SZ);
  *(float4*)&gg[4] = *(const float4*)(gp + 2 * H_SZ + 4);
  *(float4*)&go[0] = *(const float4*)(gp + 3 * H_SZ);
  *(float4*)&go[4] = *(const float4*)(gp + 3 * H_SZ + 4);
}

// ---------------- lengths ----------------
__global__ void lengths_kernel(const float* __restrict__ x, int* __restrict__ lengths) {
  int b = blockIdx.x;
  __shared__ int cnt[NTHR];
  int t = threadIdx.x;
  int local = 0;
  for (int tt = t; tt < T_SZ; tt += NTHR) {
    const float* row = x + ((size_t)b * T_SZ + tt) * NDYN_K;
    float s = 0.f;
#pragma unroll
    for (int k = 0; k < NDYN_K; ++k) s += fabsf(row[k]);
    local += (s != 0.f) ? 1 : 0;
  }
  cnt[t] = local;
  __syncthreads();
  for (int off = NTHR / 2; off > 0; off >>= 1) {
    if (t < off) cnt[t] += cnt[t + off];
    __syncthreads();
  }
  if (t == 0) lengths[b] = cnt[0];
}

// ---------------- static branch: emb -> MLP -> LN ----------------
__global__ void static_kernel(const int* __restrict__ cats, const float* __restrict__ nums,
                              const float* __restrict__ emb0, const float* __restrict__ emb1,
                              const float* __restrict__ W1, const float* __restrict__ b1,
                              const float* __restrict__ W2, const float* __restrict__ b2,
                              const float* __restrict__ nsw, const float* __restrict__ nsb,
                              float* __restrict__ outv) {
  int b = blockIdx.x, u = threadIdx.x;  // 64 threads
  __shared__ float feat[19], h1[64], red[64];
  if (u < 8)        feat[u] = emb0[(size_t)cats[b] * 8 + u];
  else if (u < 16)  feat[u] = emb1[(size_t)cats[B_SZ + b] * 8 + (u - 8)];
  else if (u < 19)  feat[u] = nums[b * 3 + (u - 16)];
  __syncthreads();
  float a = b1[u];
#pragma unroll
  for (int k = 0; k < 19; ++k) a += W1[u * 19 + k] * feat[k];
  h1[u] = fmaxf(a, 0.f);
  __syncthreads();
  float a2 = b2[u];
  for (int k = 0; k < 64; ++k) a2 += W2[u * 64 + k] * h1[k];
  a2 = fmaxf(a2, 0.f);
  red[u] = a2; __syncthreads();
  for (int off = 32; off > 0; off >>= 1) { if (u < off) red[u] += red[u + off]; __syncthreads(); }
  float m = red[0] / 64.f; __syncthreads();
  float d = a2 - m;
  red[u] = d * d; __syncthreads();
  for (int off = 32; off > 0; off >>= 1) { if (u < off) red[u] += red[u + off]; __syncthreads(); }
  float var = red[0] / 64.f;
  outv[b * 64 + u] = d * rsqrtf(var + 1e-5f) * nsw[u] + nsb[u];
}

// ---------------- encoder layer 0 (recurrent, WMMA bf16) ----------------
// A = [h(128) | x_t padded to 32], K=160 (5 chunks). W = [Whh | Wih | 0].
__global__ void enc_layer0_kernel(const float* __restrict__ x_dyn,
                                  const float* __restrict__ W_hh,   // [512,128]
                                  const float* __restrict__ W_ih,   // [512,10]
                                  const float* __restrict__ b_ih,
                                  const float* __restrict__ b_hh,
                                  const int*   __restrict__ lengths,
                                  ushort_t*    __restrict__ hs_out) // [T,B,128] bf16
{
  constexpr int NCHUNK = 5, KTOT = 160;
  extern __shared__ char smem[];
  ushort_t* Wfrag = (ushort_t*)smem;                          // 32*5*32*16 us
  ushort_t* Afrag = Wfrag + 32 * NCHUNK * 32 * 16;            // 5*32*16 us
  float*    g_buf = (float*)(Afrag + NCHUNK * 32 * 16);       // 16*512 f
  float*    biasS = g_buf + RPW * G4H;                        // 512 f

  const int tid = threadIdx.x;
  const int lane = tid & 31, wave = tid >> 5;
  const int gb0 = blockIdx.x * RPW;
  const int row = tid >> 4, jb = (tid & 15) * 8;

  for (int i = tid; i < G4H * KTOT; i += NTHR) {
    int n = i / KTOT, k = i % KTOT;
    float w = 0.f;
    if (k < H_SZ)               w = W_hh[n * H_SZ + k];
    else if (k < H_SZ + NDYN_K) w = W_ih[n * NDYN_K + (k - H_SZ)];
    int ck = k >> 5, kk = k & 31;
    int l = (n & 15) + 16 * ((kk >> 4) & 1);
    int e = kk & 15;
    Wfrag[(((size_t)(n >> 4) * NCHUNK + ck) * 32 + l) * 16 + e] = f2bf(w);
  }
  for (int i = tid; i < G4H; i += NTHR) biasS[i] = b_ih[i] + b_hh[i];
  for (int i = tid; i < NCHUNK * 32 * 16; i += NTHR) Afrag[i] = 0;

  float c_reg[8], h_reg[8];
#pragma unroll
  for (int q = 0; q < 8; ++q) { c_reg[q] = 0.f; h_reg[q] = 0.f; }
  const int my_len = lengths[gb0 + row];
  __syncthreads();

  // hoist bias accumulator seeds (constant across the 512-step loop)
  v8f bacc[4];
#pragma unroll
  for (int tt = 0; tt < 4; ++tt) {
    float bn = biasS[(wave * 4 + tt) * 16 + (lane & 15)];
#pragma unroll
    for (int r = 0; r < 8; ++r) bacc[tt][r] = bn;
  }

  for (int t = 0; t < T_SZ; ++t) {
    // stage x_t into A columns [128..138); cols 138..159 stay zero
    for (int i = tid; i < RPW * NDYN_K; i += NTHR) {
      int r = i / NDYN_K, k = i % NDYN_K;
      float xv = x_dyn[((size_t)(gb0 + r) * T_SZ + t) * NDYN_K + k];
      a_store(Afrag, r, H_SZ + k, f2bf(xv));
    }
    if (t + 1 < T_SZ && tid < RPW)
      __builtin_prefetch(&x_dyn[((size_t)(gb0 + tid) * T_SZ + (t + 1)) * NDYN_K], 0, 0);
    __syncthreads();

#pragma unroll
    for (int tt = 0; tt < 4; ++tt) {
      int tile = wave * 4 + tt;
      int n = tile * 16 + (lane & 15);
      int mb = (lane >> 4) * 8;
      v8f acc = wave_tile_gemm<NCHUNK>(Afrag, Wfrag, tile, lane, bacc[tt]);
#pragma unroll
      for (int r = 0; r < 8; ++r) g_buf[(mb + r) * G4H + n] = acc[r];
    }
    __syncthreads();

    bool valid = (t < my_len);
    float gi[8], gf[8], gg[8], go[8];
    load_gates(g_buf, row, jb, gi, gf, gg, go);
#pragma unroll
    for (int q = 0; q < 8; ++q) {
      float cn = sigf(gf[q]) * c_reg[q] + sigf(gi[q]) * tanhf(gg[q]);
      float hn = sigf(go[q]) * tanhf(cn);
      if (valid) { c_reg[q] = cn; h_reg[q] = hn; }
      ushort_t hb = f2bf(h_reg[q]);
      a_store(Afrag, row, jb + q, hb);
      hs_out[((size_t)t * B_SZ + gb0 + row) * H_SZ + jb + q] = hb;
    }
    __syncthreads();
  }
}

// ---------------- encoder layer 1 (recurrent, K=256, async-pipelined) ----------------
// A = [h1(128) | h0_t(128)], W = [Whh1 | Wih1].
// h0_t is streamed one step ahead with global_load_async_to_lds_b128 (double buffer).
__global__ void enc_layer1_kernel(const ushort_t* __restrict__ hs_in, // [T,B,128] bf16
                                  const float* __restrict__ W_hh,
                                  const float* __restrict__ W_ih,
                                  const float* __restrict__ b_ih,
                                  const float* __restrict__ b_hh,
                                  const int*   __restrict__ lengths,
                                  float*       __restrict__ hT_out)  // [B,128]
{
  constexpr int NCHUNK = 8, KTOT = 256;
  extern __shared__ char smem[];
  ushort_t* Wfrag = (ushort_t*)smem;
  ushort_t* Afrag = Wfrag + 32 * NCHUNK * 32 * 16;
  float*    g_buf = (float*)(Afrag + NCHUNK * 32 * 16);
  float*    biasS = g_buf + RPW * G4H;
  ushort_t* stage = (ushort_t*)(biasS + G4H);                 // 2 x 2048 bf16

  const int tid = threadIdx.x;
  const int lane = tid & 31, wave = tid >> 5;
  const int gb0 = blockIdx.x * RPW;
  const int row = tid >> 4, jb = (tid & 15) * 8;

  for (int i = tid; i < G4H * KTOT; i += NTHR) {
    int n = i / KTOT, k = i % KTOT;
    float w = (k < H_SZ) ? W_hh[n * H_SZ + k] : W_ih[n * H_SZ + (k - H_SZ)];
    int ck = k >> 5, kk = k & 31;
    int l = (n & 15) + 16 * ((kk >> 4) & 1);
    int e = kk & 15;
    Wfrag[(((size_t)(n >> 4) * NCHUNK + ck) * 32 + l) * 16 + e] = f2bf(w);
  }
  for (int i = tid; i < G4H; i += NTHR) biasS[i] = b_ih[i] + b_hh[i];
  for (int i = tid; i < NCHUNK * 32 * 16; i += NTHR) Afrag[i] = 0;

  // Each thread owns 16 bytes of the per-step h0 block: elements [tid*8, tid*8+8)
  // of the contiguous [16 rows x 128] bf16 slab -> (row = tid>>4, j = (tid&15)*8 + q).
  const ushort_t* gsrc0 = hs_in + (size_t)gb0 * H_SZ + tid * 8;  // + t*B*H per step
  // prologue: async copy of t=0, swizzle it, then launch copy of t=1
  async_cp16(gsrc0, stage + tid * 8);
  wait_async0();

  float c_reg[8], h_reg[8];
#pragma unroll
  for (int q = 0; q < 8; ++q) { c_reg[q] = 0.f; h_reg[q] = 0.f; }
  const int my_len = lengths[gb0 + row];
  __syncthreads();

  v8f bacc[4];
#pragma unroll
  for (int tt = 0; tt < 4; ++tt) {
    float bn = biasS[(wave * 4 + tt) * 16 + (lane & 15)];
#pragma unroll
    for (int r = 0; r < 8; ++r) bacc[tt][r] = bn;
  }
#pragma unroll
  for (int q = 0; q < 8; ++q)
    a_store(Afrag, row, H_SZ + jb + q, stage[tid * 8 + q]);
  async_cp16(gsrc0 + (size_t)B_SZ * H_SZ, stage + 2048 + tid * 8);  // t=1
  __syncthreads();

  for (int t = 0; t < T_SZ; ++t) {
#pragma unroll
    for (int tt = 0; tt < 4; ++tt) {
      int tile = wave * 4 + tt;
      int n = tile * 16 + (lane & 15);
      int mb = (lane >> 4) * 8;
      v8f acc = wave_tile_gemm<NCHUNK>(Afrag, Wfrag, tile, lane, bacc[tt]);
#pragma unroll
      for (int r = 0; r < 8; ++r) g_buf[(mb + r) * G4H + n] = acc[r];
    }
    __syncthreads();

    bool valid = (t < my_len);
    float gi[8], gf[8], gg[8], go[8];
    load_gates(g_buf, row, jb, gi, gf, gg, go);
#pragma unroll
    for (int q = 0; q < 8; ++q) {
      float cn = sigf(gf[q]) * c_reg[q] + sigf(gi[q]) * tanhf(gg[q]);
      float hn = sigf(go[q]) * tanhf(cn);
      if (valid) { c_reg[q] = cn; h_reg[q] = hn; }
      a_store(Afrag, row, jb + q, f2bf(h_reg[q]));
    }
    // pipeline: kick copy of t+2, drain copy of t+1, swizzle it for next step.
    if (t + 2 < T_SZ) {
      async_cp16(gsrc0 + (size_t)(t + 2) * B_SZ * H_SZ, stage + (t & 1) * 2048 + tid * 8);
      wait_async1();   // copy(t+1) done, copy(t+2) still in flight
      const ushort_t* sb = stage + ((t + 1) & 1) * 2048 + tid * 8;
#pragma unroll
      for (int q = 0; q < 8; ++q)
        a_store(Afrag, row, H_SZ + jb + q, sb[q]);
    } else if (t + 1 < T_SZ) {
      wait_async0();
      const ushort_t* sb = stage + ((t + 1) & 1) * 2048 + tid * 8;
#pragma unroll
      for (int q = 0; q < 8; ++q)
        a_store(Afrag, row, H_SZ + jb + q, sb[q]);
    }
    __syncthreads();
  }
#pragma unroll
  for (int q = 0; q < 8; ++q)
    hT_out[(size_t)(gb0 + row) * H_SZ + jb + q] = h_reg[q];
}

// ---------------- LN over encoder hT ----------------
__global__ void seq_ln_kernel(const float* __restrict__ hT, const float* __restrict__ w,
                              const float* __restrict__ bb, float* __restrict__ outv) {
  int b = blockIdx.x, u = threadIdx.x;  // 128 threads
  __shared__ float red[128];
  float v = hT[b * H_SZ + u];
  red[u] = v; __syncthreads();
  for (int off = 64; off > 0; off >>= 1) { if (u < off) red[u] += red[u + off]; __syncthreads(); }
  float m = red[0] / 128.f; __syncthreads();
  float d = v - m;
  red[u] = d * d; __syncthreads();
  for (int off = 64; off > 0; off >>= 1) { if (u < off) red[u] += red[u + off]; __syncthreads(); }
  float var = red[0] / 128.f;
  outv[b * H_SZ + u] = d * rsqrtf(var + 1e-5f) * w[u] + bb[u];
}

// ---------------- decoder init (torch-style flat reshape) ----------------
__global__ void dec_init_kernel(const float* __restrict__ seqv, const float* __restrict__ statv,
                                const float* __restrict__ ihW, const float* __restrict__ ihb,
                                float* __restrict__ h0, float* __restrict__ c0) {
  int b = blockIdx.x, u = threadIdx.x;  // 128 threads
  __shared__ float feat[192];
  if (u < 128) feat[u] = seqv[b * 128 + u];
  if (u < 64)  feat[128 + u] = statv[b * 64 + u];
  __syncthreads();
#pragma unroll
  for (int oo = 0; oo < 4; ++oo) {
    int o = u + oo * 128;
    float a = ihb[o];
    for (int k = 0; k < 192; ++k) a += ihW[o * 192 + k] * feat[k];
    a = tanhf(a);
    int half = o >> 8;          // 0 -> h0f, 1 -> c0f
    int kk = o & 255;
    long flat = (long)b * 256 + kk;     // index into [B,256] flattened
    int l = (int)(flat >> 15);          // / (B*H) = 32768
    int rem = (int)(flat & 32767);
    int bb2 = rem >> 7, hh = rem & 127;
    float* dst = half ? c0 : h0;
    dst[((size_t)l * B_SZ + bb2) * H_SZ + hh] = a;
  }
}

// ---------------- decoder (2-layer fused, autoregressive) ----------------
// layer0: g = inp@Wih0^T (K=2, in C-init) + h1@Whh0^T (WMMA, register-resident B frags)
// layer1: A = [h2_prev | h1], W1 = [Whh1 | Wih1] (K=256, LDS)
__global__ void decoder_kernel(const float* __restrict__ h0, const float* __restrict__ c0,
                               const float* __restrict__ Wih0, const float* __restrict__ Whh0,
                               const float* __restrict__ bih0, const float* __restrict__ bhh0,
                               const float* __restrict__ Wih1, const float* __restrict__ Whh1,
                               const float* __restrict__ bih1, const float* __restrict__ bhh1,
                               const float* __restrict__ foW, const float* __restrict__ fob,
                               float* __restrict__ out) {
  constexpr int NCK1 = 8;
  extern __shared__ char smem[];
  ushort_t* W1frag = (ushort_t*)smem;                 // 32*8*32*16 us
  ushort_t* Afrag1 = W1frag + 32 * NCK1 * 32 * 16;    // 8*32*16 us
  ushort_t* Afrag0 = Afrag1 + NCK1 * 32 * 16;         // 4*32*16 us
  float* g_buf = (float*)(Afrag0 + 4 * 32 * 16);      // 16*512 f
  float* bias0 = g_buf + RPW * G4H;                   // 512
  float* bias1 = bias0 + G4H;                         // 512
  float* wih0S = bias1 + G4H;                         // 512*2
  float* foWS  = wih0S + G4H * 2;                     // 2*128
  float* inpS  = foWS + 256;                          // 16*2
  float* pbuf  = inpS + 32;                           // 16*16*2

  const int tid = threadIdx.x;
  const int lane = tid & 31, wave = tid >> 5;
  const int gb0 = blockIdx.x * RPW;
  const int row = tid >> 4, jb = (tid & 15) * 8;

  for (int i = tid; i < G4H * 256; i += NTHR) {
    int n = i >> 8, k = i & 255;
    float w = (k < H_SZ) ? Whh1[n * H_SZ + k] : Wih1[n * H_SZ + (k - H_SZ)];
    int ck = k >> 5, kk = k & 31;
    int l = (n & 15) + 16 * ((kk >> 4) & 1);
    int e = kk & 15;
    W1frag[(((size_t)(n >> 4) * NCK1 + ck) * 32 + l) * 16 + e] = f2bf(w);
  }
  for (int i = tid; i < G4H; i += NTHR) { bias0[i] = bih0[i] + bhh0[i]; bias1[i] = bih1[i] + bhh1[i]; }
  for (int i = tid; i < G4H * OUT_K; i += NTHR) wih0S[i] = Wih0[i];
  for (int i = tid; i < OUT_K * H_SZ; i += NTHR) foWS[i] = foW[i];
  for (int i = tid; i < RPW * OUT_K; i += NTHR) inpS[i] = 0.f;

  // Whh0: register-resident B fragments for this wave's 4 N-tiles (128 VGPRs)
  v16bf w0[4][4];
#pragma unroll
  for (int tt = 0; tt < 4; ++tt) {
#pragma unroll
    for (int ck = 0; ck < 4; ++ck) {
      int n = (wave * 4 + tt) * 16 + (lane & 15);
      int kh = (lane >> 4) * 16;
#pragma unroll
      for (int e = 0; e < 16; ++e) {
        ushort_t wb = f2bf(Whh0[n * H_SZ + ck * 32 + kh + e]);
        w0[tt][ck][e] = __builtin_bit_cast(__bf16, wb);
      }
    }
  }

  float h1_reg[8], c1_reg[8], h2_reg[8], c2_reg[8];
#pragma unroll
  for (int q = 0; q < 8; ++q) {
    size_t bi = (size_t)(gb0 + row) * H_SZ + jb + q;
    h1_reg[q] = h0[bi];                       c1_reg[q] = c0[bi];
    h2_reg[q] = h0[(size_t)B_SZ * H_SZ + bi]; c2_reg[q] = c0[(size_t)B_SZ * H_SZ + bi];
    a_store(Afrag0, row, jb + q, f2bf(h1_reg[q]));
    a_store(Afrag1, row, jb + q, f2bf(h2_reg[q]));
  }
  __syncthreads();

  // hoist per-lane constants out of the step loop
  v8f b1acc[4];
  float b0n[4], wx0a[4], wx1a[4];
#pragma unroll
  for (int tt = 0; tt < 4; ++tt) {
    int n = (wave * 4 + tt) * 16 + (lane & 15);
    b0n[tt] = bias0[n];
    wx0a[tt] = wih0S[n * 2 + 0];
    wx1a[tt] = wih0S[n * 2 + 1];
    float bn1 = bias1[n];
#pragma unroll
    for (int r = 0; r < 8; ++r) b1acc[tt][r] = bn1;
  }

  for (int t = 0; t < FSTEP; ++t) {
    // ---- layer 0 GEMM ----
#pragma unroll
    for (int tt = 0; tt < 4; ++tt) {
      int tile = wave * 4 + tt;
      int n = tile * 16 + (lane & 15);
      int mb = (lane >> 4) * 8;
      v8f acc;
#pragma unroll
      for (int r = 0; r < 8; ++r)
        acc[r] = b0n[tt] + inpS[(mb + r) * 2] * wx0a[tt] + inpS[(mb + r) * 2 + 1] * wx1a[tt];
#pragma unroll
      for (int ck = 0; ck < 4; ++ck) {
        v16bf a = *(const v16bf*)(Afrag0 + (ck * 32 + lane) * 16);
        acc = __builtin_amdgcn_wmma_f32_16x16x32_bf16(false, a, false, w0[tt][ck],
                                                      (short)0, acc, false, false);
      }
#pragma unroll
      for (int r = 0; r < 8; ++r) g_buf[(mb + r) * G4H + n] = acc[r];
    }
    __syncthreads();
    // ---- cell 0 ----
    {
      float gi[8], gf[8], gg[8], go[8];
      load_gates(g_buf, row, jb, gi, gf, gg, go);
#pragma unroll
      for (int q = 0; q < 8; ++q) {
        c1_reg[q] = sigf(gf[q]) * c1_reg[q] + sigf(gi[q]) * tanhf(gg[q]);
        h1_reg[q] = sigf(go[q]) * tanhf(c1_reg[q]);
        ushort_t hb = f2bf(h1_reg[q]);
        a_store(Afrag0, row, jb + q, hb);            // next step's layer-0 GEMM
        a_store(Afrag1, row, H_SZ + jb + q, hb);     // this step's layer-1 input
      }
    }
    __syncthreads();
    // ---- layer 1 GEMM ----
#pragma unroll
    for (int tt = 0; tt < 4; ++tt) {
      int tile = wave * 4 + tt;
      int n = tile * 16 + (lane & 15);
      int mb = (lane >> 4) * 8;
      v8f acc = wave_tile_gemm<NCK1>(Afrag1, W1frag, tile, lane, b1acc[tt]);
#pragma unroll
      for (int r = 0; r < 8; ++r) g_buf[(mb + r) * G4H + n] = acc[r];
    }
    __syncthreads();
    // ---- cell 1 + output-projection partials ----
    {
      float gi[8], gf[8], gg[8], go[8];
      load_gates(g_buf, row, jb, gi, gf, gg, go);
      float p0 = 0.f, p1 = 0.f;
#pragma unroll
      for (int q = 0; q < 8; ++q) {
        c2_reg[q] = sigf(gf[q]) * c2_reg[q] + sigf(gi[q]) * tanhf(gg[q]);
        h2_reg[q] = sigf(go[q]) * tanhf(c2_reg[q]);
        a_store(Afrag1, row, jb + q, f2bf(h2_reg[q]));
        p0 += h2_reg[q] * foWS[jb + q];
        p1 += h2_reg[q] * foWS[H_SZ + jb + q];
      }
      pbuf[(row * 16 + (tid & 15)) * 2 + 0] = p0;
      pbuf[(row * 16 + (tid & 15)) * 2 + 1] = p1;
    }
    __syncthreads();
    if (tid < RPW * OUT_K) {
      int r = tid >> 1, o = tid & 1;
      float s = fob[o];
#pragma unroll
      for (int f = 0; f < 16; ++f) s += pbuf[(r * 16 + f) * 2 + o];
      inpS[r * 2 + o] = s;
      out[((size_t)(gb0 + r) * FSTEP + t) * OUT_K + o] = s;
    }
    __syncthreads();
  }
}

// ---------------- host ----------------
extern "C" void kernel_launch(void* const* d_in, const int* in_sizes, int n_in,
                              void* d_out, int out_size, void* d_ws, size_t ws_size,
                              hipStream_t stream) {
  (void)in_sizes; (void)n_in; (void)out_size; (void)ws_size;
  const float* x_dyn  = (const float*)d_in[0];
  const int*   cats   = (const int*)d_in[1];
  const float* nums   = (const float*)d_in[2];
  const float* emb0   = (const float*)d_in[3];
  const float* emb1   = (const float*)d_in[4];
  const float* eW_ih0 = (const float*)d_in[5];
  const float* eW_hh0 = (const float*)d_in[6];
  const float* eb_ih0 = (const float*)d_in[7];
  const float* eb_hh0 = (const float*)d_in[8];
  const float* eW_ih1 = (const float*)d_in[9];
  const float* eW_hh1 = (const float*)d_in[10];
  const float* eb_ih1 = (const float*)d_in[11];
  const float* eb_hh1 = (const float*)d_in[12];
  const float* dW_ih0 = (const float*)d_in[13];
  const float* dW_hh0 = (const float*)d_in[14];
  const float* db_ih0 = (const float*)d_in[15];
  const float* db_hh0 = (const float*)d_in[16];
  const float* dW_ih1 = (const float*)d_in[17];
  const float* dW_hh1 = (const float*)d_in[18];
  const float* db_ih1 = (const float*)d_in[19];
  const float* db_hh1 = (const float*)d_in[20];
  const float* fs_W1  = (const float*)d_in[21];
  const float* fs_b1  = (const float*)d_in[22];
  const float* fs_W2  = (const float*)d_in[23];
  const float* fs_b2  = (const float*)d_in[24];
  const float* ne_w   = (const float*)d_in[25];
  const float* ne_b   = (const float*)d_in[26];
  const float* ns_w   = (const float*)d_in[27];
  const float* ns_b   = (const float*)d_in[28];
  const float* ih_W   = (const float*)d_in[29];
  const float* ih_b   = (const float*)d_in[30];
  const float* fo_W   = (const float*)d_in[31];
  const float* fo_b   = (const float*)d_in[32];

  char* ws = (char*)d_ws;
  size_t off = 0;
  int*      lengths    = (int*)(ws + off);       off += 1024;
  float*    static_vec = (float*)(ws + off);     off += (size_t)B_SZ * 64 * 4;
  float*    hT         = (float*)(ws + off);     off += (size_t)B_SZ * H_SZ * 4;
  float*    seq_vec    = (float*)(ws + off);     off += (size_t)B_SZ * H_SZ * 4;
  float*    h0         = (float*)(ws + off);     off += (size_t)2 * B_SZ * H_SZ * 4;
  float*    c0         = (float*)(ws + off);     off += (size_t)2 * B_SZ * H_SZ * 4;
  ushort_t* hs0        = (ushort_t*)(ws + off);  off += (size_t)T_SZ * B_SZ * H_SZ * 2;

  const int SMEM0 = (32 * 5 * 32 * 16 + 5 * 32 * 16) * 2 + (RPW * G4H + G4H) * 4;
  const int SMEM1 = (32 * 8 * 32 * 16 + 8 * 32 * 16) * 2 + (RPW * G4H + G4H) * 4 +
                    2 * 2048 * 2;
  const int SMEMD = (32 * 8 * 32 * 16 + 8 * 32 * 16 + 4 * 32 * 16) * 2 +
                    (RPW * G4H + G4H + G4H + G4H * 2 + 256 + 32 + 512) * 4;
  (void)hipFuncSetAttribute((const void*)enc_layer0_kernel,
                            hipFuncAttributeMaxDynamicSharedMemorySize, SMEM0);
  (void)hipFuncSetAttribute((const void*)enc_layer1_kernel,
                            hipFuncAttributeMaxDynamicSharedMemorySize, SMEM1);
  (void)hipFuncSetAttribute((const void*)decoder_kernel,
                            hipFuncAttributeMaxDynamicSharedMemorySize, SMEMD);

  lengths_kernel<<<B_SZ, NTHR, 0, stream>>>(x_dyn, lengths);
  static_kernel<<<B_SZ, 64, 0, stream>>>(cats, nums, emb0, emb1, fs_W1, fs_b1,
                                         fs_W2, fs_b2, ns_w, ns_b, static_vec);
  enc_layer0_kernel<<<B_SZ / RPW, NTHR, SMEM0, stream>>>(
      x_dyn, eW_hh0, eW_ih0, eb_ih0, eb_hh0, lengths, hs0);
  enc_layer1_kernel<<<B_SZ / RPW, NTHR, SMEM1, stream>>>(
      hs0, eW_hh1, eW_ih1, eb_ih1, eb_hh1, lengths, hT);
  seq_ln_kernel<<<B_SZ, H_SZ, 0, stream>>>(hT, ne_w, ne_b, seq_vec);
  dec_init_kernel<<<B_SZ, H_SZ, 0, stream>>>(seq_vec, static_vec, ih_W, ih_b, h0, c0);
  decoder_kernel<<<B_SZ / RPW, NTHR, SMEMD, stream>>>(
      h0, c0, dW_ih0, dW_hh0, db_ih0, db_hh0, dW_ih1, dW_hh1, db_ih1, db_hh1,
      fo_W, fo_b, (float*)d_out);
}
